// Self_Attn_52381421142597
// MI455X (gfx1250) — matrix operand order repaired and checked by reading, not compiled, not run on hardware
//
#include <hip/hip_runtime.h>

// ---------------------------------------------------------------------------
// Problem constants
// ---------------------------------------------------------------------------
static constexpr int Bb = 8;      // batch
static constexpr int Cc = 256;    // channels
static constexpr int C8 = 32;     // q/k channels
static constexpr int NN = 2304;   // H*W = 48*48
static constexpr int KP = 36;     // LDS row pitch (32 + 4 pad elems, 72B)
static constexpr int NCH = NN / 32;  // 72 j-chunks

typedef __attribute__((ext_vector_type(16))) __bf16 bf16x16;
typedef __attribute__((ext_vector_type(8)))  __bf16 bf16x8;
typedef __attribute__((ext_vector_type(8)))  float  v8f;
typedef __attribute__((ext_vector_type(4)))  unsigned int uint32x4;
typedef __attribute__((ext_vector_type(8)))  int    int32x8;
typedef __attribute__((ext_vector_type(4)))  int    int32x4;

__device__ __forceinline__ v8f wmma_bf16(bf16x16 a, bf16x16 b, v8f c) {
    return __builtin_amdgcn_wmma_f32_16x16x32_bf16(
        false, a, false, b, (short)0, c, false, false);
}

// A fragment (16x32 MxK, 16-bit): lane holds row m = L&15;
// element e -> K = ((e&8)?16:0) + (L>>4)*8 + (e&7). base = row-major [16][KP].
__device__ __forceinline__ bf16x16 load_A_frag(const __bf16* base) {
    const int lane = threadIdx.x & 31;
    const int m = lane & 15, hf = lane >> 4;
    bf16x16 a;
#pragma unroll
    for (int e = 0; e < 16; ++e) {
        int k = ((e & 8) ? 16 : 0) + hf * 8 + (e & 7);
        a[e] = base[m * KP + k];
    }
    return a;
}

// B fragment (32x16 KxN, 16-bit): lane holds col n = L&15; K = ((e&8)?16:0) + (L>>4)*8 + (e&7).
// base = [n][k] tile with pitch KP.
__device__ __forceinline__ bf16x16 load_B_frag(const __bf16* base) {
    const int lane = threadIdx.x & 31;
    const int n = lane & 15, hf = lane >> 4;
    bf16x16 b;
#pragma unroll
    for (int e = 0; e < 16; ++e) {
        b[e] = base[n * KP + ((e & 8) ? 16 : 0) + hf * 8 + (e & 7)];
    }
    return b;
}

__device__ __forceinline__ float half_reduce_max(float v) {
    v = fmaxf(v, __shfl_xor(v, 1, 32));
    v = fmaxf(v, __shfl_xor(v, 2, 32));
    v = fmaxf(v, __shfl_xor(v, 4, 32));
    v = fmaxf(v, __shfl_xor(v, 8, 32));
    return v;
}
__device__ __forceinline__ float half_reduce_sum(float v) {
    v += __shfl_xor(v, 1, 32);
    v += __shfl_xor(v, 2, 32);
    v += __shfl_xor(v, 4, 32);
    v += __shfl_xor(v, 8, 32);
    return v;
}

// ---------------------------------------------------------------------------
// Tensor Data Mover: async copy of a [Cc rows x 32 cols] bf16 tile
// (row stride NN elems) from global into LDS with pitch KP (=32+4) elems.
// D# built per CDNA5 ISA 8.3/8.4:
//   group0: count=1 | lds_addr | global_addr(57b) | type=2
//   group1: data_size=2B, pad_enable, pad_interval=16DW(64B=row),
//           pad_amount=2DW(8B=4 elems), tensor=[NN x Cc], tile=[32 x Cc],
//           dim0_stride=NN, workgroup_mask=0 (not in cluster)
// ---------------------------------------------------------------------------
__device__ __forceinline__ void tdm_load_v_chunk(const __bf16* gsrc,
                                                 unsigned int lds_addr) {
    const unsigned long long ga = (unsigned long long)(uintptr_t)gsrc;
    uint32x4 g0;
    g0[0] = 1u;                                    // count=1, user mode
    g0[1] = lds_addr;                              // LDS byte address
    g0[2] = (unsigned int)ga;                      // global addr [31:0]
    g0[3] = (unsigned int)((ga >> 32) & 0x1FFFFFFu) | (2u << 30);  // [56:32]|type=2
    int32x8 g1;
    g1[0] = (int)((1u << 16)        // data_size = 2 bytes
                | (1u << 20)        // pad_enable
                | (3u << 22)        // pad_interval: 16 DWORDs (= one 64B row)
                | (1u << 25));      // pad_amount: 2 DWORDs (= 4 bf16 -> pitch 36)
    g1[1] = (int)(((unsigned)NN & 0xFFFFu) << 16);                 // dim0 lo16
    g1[2] = (int)((((unsigned)NN >> 16) & 0xFFFFu)
                | (((unsigned)Cc & 0xFFFFu) << 16));               // dim0 hi | dim1 lo
    g1[3] = (int)((((unsigned)Cc >> 16) & 0xFFFFu) | (32u << 16)); // dim1 hi | tile_dim0
    g1[4] = (int)((unsigned)Cc);    // tile_dim1 = Cc, tile_dim2 = 0
    g1[5] = (int)((unsigned)NN);    // tensor_dim0_stride lo32
    g1[6] = 0;                      // stride hi | dim1_stride lo (unused, 2D tile)
    g1[7] = 0;
    int32x4 z4 = {0, 0, 0, 0};
#if defined(__clang_major__) && (__clang_major__ >= 23)
    int32x8 z8 = {0, 0, 0, 0, 0, 0, 0, 0};
    __builtin_amdgcn_tensor_load_to_lds(g0, g1, z4, z4, z8, 0);
#else
    __builtin_amdgcn_tensor_load_to_lds(g0, g1, z4, z4, 0);
#endif
}

// ---------------------------------------------------------------------------
// 1x1 conv as WMMA GEMM: out[b,o,n] = sum_c W[o,c]*X[b,c,n] + bias[o]
// grid = (N/64, O/16, B), one wave; 16(o) x 64(n) tile per block.
// ---------------------------------------------------------------------------
__global__ __launch_bounds__(32) void proj_kernel(
    const float* __restrict__ X, const float* __restrict__ W,
    const float* __restrict__ bias, __bf16* __restrict__ out, int O)
{
    __shared__ __align__(16) __bf16 aT[16 * KP];      // [m][k]  W tile
    __shared__ __align__(16) __bf16 bT[4][16 * KP];   // [n][k]  X tiles (transposed)
    const int lane = threadIdx.x;
    const int n0 = blockIdx.x * 64;
    const int o0 = blockIdx.y * 16;
    const int b  = blockIdx.z;
    const int ln = lane & 15, hf = lane >> 4;

    // per-lane staging geometry (contiguous runs, hoisted bases)
    const int am = lane >> 1;               // A: row m, 16 contiguous k
    const int ah = (lane & 1) * 16;
    const float* wsrc = W + (size_t)(o0 + am) * Cc + ah;
    __bf16* adst = &aT[am * KP + ah];
    const float* xsrc = X + ((size_t)b * Cc + lane) * NN + n0;  // B: row c=lane, 64 n
    __bf16* bdst = &bT[0][lane];

    v8f acc[4];
    const v8f zero = {0.f, 0.f, 0.f, 0.f, 0.f, 0.f, 0.f, 0.f};
#pragma unroll
    for (int t = 0; t < 4; ++t) acc[t] = zero;

    for (int kc = 0; kc < Cc / 32; ++kc) {
        __syncthreads();
        {   // stage A: 16 contiguous floats -> bf16
            const float4* wp = (const float4*)(wsrc + kc * 32);
#pragma unroll
            for (int q = 0; q < 4; ++q) {
                float4 f = wp[q];
                adst[q * 4 + 0] = (__bf16)f.x;
                adst[q * 4 + 1] = (__bf16)f.y;
                adst[q * 4 + 2] = (__bf16)f.z;
                adst[q * 4 + 3] = (__bf16)f.w;
            }
        }
        {   // stage B: 64 contiguous floats of row c -> transposed bf16
            const float4* xp = (const float4*)(xsrc + (size_t)kc * 32 * NN);
#pragma unroll
            for (int q = 0; q < 16; ++q) {
                float4 f = xp[q];
                int n = q * 4;
#pragma unroll
                for (int cmp = 0; cmp < 4; ++cmp) {
                    float v = (cmp == 0) ? f.x : (cmp == 1) ? f.y : (cmp == 2) ? f.z : f.w;
                    int nn = n + cmp;
                    bdst[(nn >> 4) * (16 * KP) + (nn & 15) * KP] = (__bf16)v;
                }
            }
        }
        __syncthreads();
        bf16x16 af = load_A_frag(aT);
#pragma unroll
        for (int t = 0; t < 4; ++t)
            acc[t] = wmma_bf16(af, load_B_frag(&bT[t][0]), acc[t]);
    }
#pragma unroll
    for (int r = 0; r < 8; ++r) {
        int o = o0 + r + 8 * hf;
        float bv = bias[o];
        __bf16* orow = out + ((size_t)b * O + o) * NN + n0 + ln;
#pragma unroll
        for (int t = 0; t < 4; ++t)
            orow[t * 16] = (__bf16)(acc[t][r] + bv);
    }
}

// ---------------------------------------------------------------------------
// Fused flash attention with TDM double-buffered V staging.
// out[b,c,i] = g * sum_j softmax_j(q^T k)[i,j] * V[c,j] + res[b,c,i]
// Block = 4 waves (wave w owns rows [i0,i0+16)); grid = (N/64, B).
// ---------------------------------------------------------------------------
__global__ __launch_bounds__(128) void flash_kernel(
    const __bf16* __restrict__ Q, const __bf16* __restrict__ K,
    const __bf16* __restrict__ V, const float* __restrict__ res,
    const float* __restrict__ gamma, float* __restrict__ out)
{
    __shared__ __align__(16) __bf16 kTt[32 * KP];        // [j][c]  K^T chunk
    __shared__ __align__(16) __bf16 vbuf[2][Cc * KP];    // [c][j]  V chunks (ping/pong)
    __shared__ __align__(16) __bf16 pT[4][16 * KP];      // per-wave P staging [m][j]

    const int tid  = threadIdx.x;
    const int lane = tid & 31;
    const int w    = tid >> 5;
    const int b    = blockIdx.y;
    const int i0   = (blockIdx.x * 4 + w) * 16;
    const int ln = lane & 15, hf = lane >> 4;

    const __bf16* Qb = Q + (size_t)b * C8 * NN;
    const __bf16* Kb = K + (size_t)b * C8 * NN;
    const __bf16* Vb = V + (size_t)b * Cc * NN;

    // Q fragment: A[m=i][k=c] = Q[c][i0+m]
    bf16x16 qa;
#pragma unroll
    for (int e = 0; e < 16; ++e) {
        int kc = ((e & 8) ? 16 : 0) + hf * 8 + (e & 7);
        qa[e] = Qb[(size_t)kc * NN + i0 + ln];
    }

    // K staging geometry: thread owns c = tid&31, 8 contiguous j's
    const int kcc = tid & 31;
    const int kjh = tid >> 5;  // 0..3
    const __bf16* ksrc = Kb + (size_t)kcc * NN + kjh * 8;
    __bf16* kdst = &kTt[(kjh * 8) * KP + kcc];

    const v8f zero = {0.f, 0.f, 0.f, 0.f, 0.f, 0.f, 0.f, 0.f};
    float rm[8], rl[8];
    v8f acc[16];
#pragma unroll
    for (int r = 0; r < 8; ++r) { rm[r] = -3.0e38f; rl[r] = 0.f; }
#pragma unroll
    for (int t = 0; t < 16; ++t) acc[t] = zero;

    // prime the TDM pipeline with chunk 0
    if (w == 0)
        tdm_load_v_chunk(Vb, (unsigned int)(uintptr_t)&vbuf[0][0]);

    for (int jc = 0; jc < NCH; ++jc) {
        const int cur = jc & 1;
        __syncthreads();  // all reads of vbuf[cur^1] / kTt from prev chunk done
        if (w == 0 && jc + 1 < NCH)
            tdm_load_v_chunk(Vb + (size_t)(jc + 1) * 32,
                             (unsigned int)(uintptr_t)&vbuf[cur ^ 1][0]);
        // stage K^T chunk (vector load, transposed scalar stores)
        {
            bf16x8 kv = *(const bf16x8*)(ksrc + jc * 32);
#pragma unroll
            for (int jj = 0; jj < 8; ++jj) kdst[jj * KP] = kv[jj];
        }
        // wait for this chunk's V DMA (keep next one in flight)
        if (jc + 1 < NCH) __builtin_amdgcn_s_wait_tensorcnt(1);
        else              __builtin_amdgcn_s_wait_tensorcnt(0);
        __syncthreads();

        // S tiles: D[i][j] = sum_c Q[c,i] K[c,j]
        bf16x16 kb0 = load_B_frag(&kTt[0]);
        bf16x16 kb1 = load_B_frag(&kTt[16 * KP]);
        v8f s0 = wmma_bf16(qa, kb0, zero);
        v8f s1 = wmma_bf16(qa, kb1, zero);

        // online softmax over this 32-column chunk
#pragma unroll
        for (int r = 0; r < 8; ++r) {
            float tmax = half_reduce_max(fmaxf(s0[r], s1[r]));
            float nm = fmaxf(rm[r], tmax);
            float scale = __expf(rm[r] - nm);
            float p0 = __expf(s0[r] - nm);
            float p1 = __expf(s1[r] - nm);
            rl[r] = rl[r] * scale + half_reduce_sum(p0 + p1);
            rm[r] = nm;
            s0[r] = p0; s1[r] = p1;
#pragma unroll
            for (int t = 0; t < 16; ++t) acc[t][r] *= scale;
        }

        // D-layout P -> LDS -> A-layout fragment
        __bf16* pw = pT[w];
#pragma unroll
        for (int r = 0; r < 8; ++r) {
            pw[(r + 8 * hf) * KP + ln]      = (__bf16)s0[r];
            pw[(r + 8 * hf) * KP + 16 + ln] = (__bf16)s1[r];
        }
        __syncthreads();
        bf16x16 pa = load_A_frag(pw);
        const __bf16* vchunk = &vbuf[cur][0];
#pragma unroll
        for (int t = 0; t < 16; ++t) {
            bf16x16 vbf = load_B_frag(vchunk + t * 16 * KP);
            acc[t] = wmma_bf16(pa, vbf, acc[t]);
        }
    }

    const float g = gamma[0];
    float rinv[8];
#pragma unroll
    for (int r = 0; r < 8; ++r) rinv[r] = 1.0f / rl[r];
#pragma unroll
    for (int t = 0; t < 16; ++t) {
#pragma unroll
        for (int r = 0; r < 8; ++r) {
            int c = t * 16 + ln;
            int i = i0 + r + 8 * hf;
            size_t idx = ((size_t)b * Cc + c) * NN + i;
            out[idx] = g * (acc[t][r] * rinv[r]) + res[idx];
        }
    }
}

// ---------------------------------------------------------------------------
// Host launcher
// ---------------------------------------------------------------------------
extern "C" void kernel_launch(void* const* d_in, const int* in_sizes, int n_in,
                              void* d_out, int out_size, void* d_ws, size_t ws_size,
                              hipStream_t stream) {
    (void)in_sizes; (void)n_in; (void)out_size; (void)ws_size;
    const float* x   = (const float*)d_in[0];
    const float* y   = (const float*)d_in[1];
    const float* wqx = (const float*)d_in[2];
    const float* bqx = (const float*)d_in[3];
    const float* wkx = (const float*)d_in[4];
    const float* bkx = (const float*)d_in[5];
    const float* wvx = (const float*)d_in[6];
    const float* bvx = (const float*)d_in[7];
    const float* wqy = (const float*)d_in[8];
    const float* bqy = (const float*)d_in[9];
    const float* wky = (const float*)d_in[10];
    const float* bky = (const float*)d_in[11];
    const float* wvy = (const float*)d_in[12];
    const float* bvy = (const float*)d_in[13];
    const float* gx  = (const float*)d_in[14];
    const float* gy  = (const float*)d_in[15];

    char* ws = (char*)d_ws;
    const size_t qkBytes = (size_t)Bb * C8 * NN * sizeof(__bf16);
    const size_t vBytes  = (size_t)Bb * Cc * NN * sizeof(__bf16);
    __bf16* q1 = (__bf16*)(ws);
    __bf16* k1 = (__bf16*)(ws + 1 * qkBytes);
    __bf16* q2 = (__bf16*)(ws + 2 * qkBytes);
    __bf16* k2 = (__bf16*)(ws + 3 * qkBytes);
    __bf16* v1 = (__bf16*)(ws + 4 * qkBytes);
    __bf16* v2 = (__bf16*)(ws + 4 * qkBytes + vBytes);

    const dim3 pblk(32);
    proj_kernel<<<dim3(NN / 64,  2, Bb), pblk, 0, stream>>>(x, wqx, bqx, q1, C8);
    proj_kernel<<<dim3(NN / 64,  2, Bb), pblk, 0, stream>>>(x, wkx, bkx, k1, C8);
    proj_kernel<<<dim3(NN / 64, 16, Bb), pblk, 0, stream>>>(x, wvx, bvx, v1, Cc);
    proj_kernel<<<dim3(NN / 64,  2, Bb), pblk, 0, stream>>>(y, wqy, bqy, q2, C8);
    proj_kernel<<<dim3(NN / 64,  2, Bb), pblk, 0, stream>>>(y, wky, bky, k2, C8);
    proj_kernel<<<dim3(NN / 64, 16, Bb), pblk, 0, stream>>>(y, wvy, bvy, v2, Cc);

    float* out1 = (float*)d_out;
    float* out2 = out1 + (size_t)Bb * Cc * NN;
    flash_kernel<<<dim3(NN / 64, Bb), 128, 0, stream>>>(q2, k2, v1, x, gx, out1);
    flash_kernel<<<dim3(NN / 64, Bb), 128, 0, stream>>>(q1, k1, v2, y, gy, out2);
}
